// GQAttention_2542620639500
// MI455X (gfx1250) — matrix operand order repaired and checked by reading, not compiled, not run on hardware
//
#include <hip/hip_runtime.h>

typedef unsigned short u16;
typedef __attribute__((ext_vector_type(16))) __bf16 v16bf;
typedef __attribute__((ext_vector_type(8)))  float  v8f;

#define S_LEN 2048
#define HIDN  2048
#define NHQ   16
#define NKVH  8
#define DHEAD 128
#define NB    2

#ifndef __has_builtin
#define __has_builtin(x) 0
#endif
#if __has_builtin(__builtin_amdgcn_global_load_async_to_lds_b128) && \
    __has_builtin(__builtin_amdgcn_s_wait_asynccnt)
#define HAVE_ASYNC_LDS 1
#else
#define HAVE_ASYNC_LDS 0
#endif

#if HAVE_ASYNC_LDS
// builtin signature: (v4i32 addrspace(1)*, v4i32 addrspace(3)*, imm int, imm int)
typedef int v4i_t __attribute__((__vector_size__(16)));
typedef __attribute__((address_space(1))) v4i_t* gptr128_t;
typedef __attribute__((address_space(3))) v4i_t* lptr128_t;
#define TO_GLOBAL(p) ((gptr128_t)(v4i_t*)(void*)(size_t)(p))
#define TO_LDS(p)    ((lptr128_t)(v4i_t*)(void*)(p))
#endif

union F16 { v16bf v; unsigned u[8]; };

__device__ __forceinline__ float bf2f(u16 h) {
  union { unsigned u; float f; } x; x.u = ((unsigned)h) << 16; return x.f;
}
__device__ __forceinline__ u16 f2bf(float f) {
  union { unsigned u; float f; } x; x.f = f;
  unsigned r = x.u + 0x7FFFu + ((x.u >> 16) & 1u);
  return (u16)(r >> 16);
}
// A-matrix (16x32 bf16) per-lane K offset for VGPR v (ISA 7.12.2 table)
__device__ __forceinline__ int a_koff(int half, int v) {
  return half * 8 + ((v & 4) ? 16 : 0) + (v & 3) * 2;
}

// ---------------------------------------------------------------- convert
__global__ void cvt_f32_to_bf16(const float* __restrict__ in,
                                u16* __restrict__ out, long n) {
  long i = (long)blockIdx.x * blockDim.x + threadIdx.x;
  long stride = (long)gridDim.x * blockDim.x;
  for (; i < n; i += stride) out[i] = f2bf(in[i]);
}

// ---------------------------------------------------------------- RoPE
// x: [rows, DHEAD] bf16, rows = B*H*S (position = row % S). scale folds 1/sqrt(D) for Q.
__global__ void rope_bf16(u16* __restrict__ x, long rows, float scale) {
  long n = rows * (DHEAD / 2);
  long idx = (long)blockIdx.x * blockDim.x + threadIdx.x;
  long stride = (long)gridDim.x * blockDim.x;
  for (; idx < n; idx += stride) {
    long r = idx >> 6;            // / (DHEAD/2)
    int  i = (int)(idx & 63);
    int  s = (int)(r % S_LEN);
    float freq = __powf(1000000.0f, -((float)(2 * i) / (float)DHEAD));
    float ang = (float)s * freq;
    float sn, cs; __sincosf(ang, &sn, &cs);
    u16* p = x + (size_t)r * DHEAD;
    float x1 = bf2f(p[i]), x2 = bf2f(p[i + 64]);
    p[i]      = f2bf((x1 * cs - x2 * sn) * scale);
    p[i + 64] = f2bf((x2 * cs + x1 * sn) * scale);
  }
}

// ---------------------------------------------------------------- GEMM
// C[M,N] = A[M,K](bf16) @ B[K,N](bf16) (+bias). mode 0: bf16 out in [B,h,S,D]
// layout (nh heads). mode 1: f32 out row-major. Block tile 128x128, K-step 32.
__global__ __launch_bounds__(256) void gemm_bf16(
    const u16* __restrict__ A, const u16* __restrict__ Bm,
    const float* __restrict__ bias,
    float* __restrict__ outF, u16* __restrict__ outB,
    int M, int N, int K, int mode, int nh) {
  __shared__ __align__(16) u16 As[128 * 40];   // [m][k], stride 40 (bank spread)
  __shared__ __align__(16) u16 Bt[128 * 40];   // [n][k] (transposed), stride 40

  const int t = threadIdx.x;
  const int w = t >> 5, lane = t & 31, half = lane >> 4, ln = lane & 15;
  const int bx = blockIdx.x, by = blockIdx.y;
  const int wr = w >> 1;            // wave row (0..3) -> 32 rows each
  const int wcl = (w & 1) * 64;     // wave col base within block tile

  v8f acc[2][4];
#pragma unroll
  for (int i = 0; i < 2; ++i)
#pragma unroll
    for (int j = 0; j < 4; ++j)
#pragma unroll
      for (int e = 0; e < 8; ++e) acc[i][j][e] = 0.0f;

  const int arow = t >> 1, aseg = (t & 1) * 16;   // A: 128 rows x 32 k
  const int brow = t >> 3, bseg = (t & 7) * 16;   // B: 32 k rows x 128 n

  for (int kk = 0; kk < K; kk += 32) {
    { // stage A tile (row-major) -- async DMA into LDS when available
      const u16* g = A + (size_t)(by * 128 + arow) * K + kk + aseg;
      u16* l = &As[arow * 40 + aseg];
#if HAVE_ASYNC_LDS
      __builtin_amdgcn_global_load_async_to_lds_b128(TO_GLOBAL(g),     TO_LDS(l),     0, 0);
      __builtin_amdgcn_global_load_async_to_lds_b128(TO_GLOBAL(g + 8), TO_LDS(l + 8), 0, 0);
#else
      const uint4* src = (const uint4*)g;
      uint4 a0 = src[0], a1 = src[1];
      *(uint4*)l       = a0;
      *(uint4*)(l + 8) = a1;
#endif
    }
    { // stage B tile transposed: Bt[n][k]
      const u16* src = Bm + (size_t)(kk + brow) * N + bx * 128 + bseg;
      union { uint4 q; u16 s[8]; } b0, b1;
      b0.q = *(const uint4*)src;
      b1.q = *(const uint4*)(src + 8);
#pragma unroll
      for (int e = 0; e < 8; ++e) {
        Bt[(bseg + e) * 40 + brow]     = b0.s[e];
        Bt[(bseg + 8 + e) * 40 + brow] = b1.s[e];
      }
    }
    if (kk + 32 < K) { // gfx1250 global_prefetch of next tiles
      __builtin_prefetch(A + (size_t)(by * 128 + arow) * K + kk + 32 + aseg, 0, 1);
      __builtin_prefetch(Bm + (size_t)(kk + 32 + brow) * N + bx * 128 + bseg, 0, 1);
    }
#if HAVE_ASYNC_LDS
    __builtin_amdgcn_s_wait_asynccnt(0);
#endif
    __syncthreads();

    F16 af[2], bf[4];
#pragma unroll
    for (int ms = 0; ms < 2; ++ms)
#pragma unroll
      for (int v = 0; v < 8; ++v)
        af[ms].u[v] = *(const unsigned*)&As[(wr * 32 + ms * 16 + ln) * 40 + a_koff(half, v)];
#pragma unroll
    for (int ns = 0; ns < 4; ++ns)
#pragma unroll
      for (int v = 0; v < 8; ++v)
        bf[ns].u[v] = *(const unsigned*)&Bt[(wcl + ns * 16 + ln) * 40 + a_koff(half, v)];
#pragma unroll
    for (int ms = 0; ms < 2; ++ms)
#pragma unroll
      for (int ns = 0; ns < 4; ++ns)
        acc[ms][ns] = __builtin_amdgcn_wmma_f32_16x16x32_bf16(
            false, af[ms].v, false, bf[ns].v, (short)0, acc[ms][ns], false, false);
    __syncthreads();
  }

  // epilogue: C/D layout lane (r -> M = r + 8*half, N = ln)
#pragma unroll
  for (int ms = 0; ms < 2; ++ms) {
#pragma unroll
    for (int ns = 0; ns < 4; ++ns) {
      const int ng = bx * 128 + wcl + ns * 16 + ln;
      const float bvv = bias ? bias[ng] : 0.0f;
#pragma unroll
      for (int r = 0; r < 8; ++r) {
        const int mg = by * 128 + wr * 32 + ms * 16 + r + 8 * half;
        const float val = acc[ms][ns][r] + bvv;
        if (mode == 0) {
          const int b = mg / S_LEN, s = mg - b * S_LEN;
          const int h = ng / DHEAD, d = ng - h * DHEAD;
          outB[(((size_t)b * nh + h) * S_LEN + s) * DHEAD + d] = f2bf(val);
        } else {
          outF[(size_t)mg * N + ng] = val;
        }
      }
    }
  }
}

// ---------------------------------------------------------------- attention
#define KSTR 136   // Ks row stride (elems): 272B, 16B aligned, bank-spread
#define VSTR 72    // Vt row stride
#define PSTR 72    // P scratch row stride

__global__ __launch_bounds__(256) void attn_fwd(
    const u16* __restrict__ Q, const u16* __restrict__ K,
    const u16* __restrict__ V, u16* __restrict__ ctx) {
  __shared__ __align__(16) u16 Ks[64 * KSTR];     // K tile [key][d]
  __shared__ __align__(16) u16 Vt[128 * VSTR];    // V tile transposed [d][key]
  __shared__ __align__(16) u16 Pb[8][16 * PSTR];  // per-wave P scratch [m][key]

  const int t = threadIdx.x;
  const int w = t >> 5, lane = t & 31, half = lane >> 4, ln = lane & 15;
  const int qt = blockIdx.x & 15;                 // q tile (128 rows)
  const int hh = (blockIdx.x >> 4) & (NHQ - 1);   // q head
  const int b  = blockIdx.x >> 8;                 // batch
  const int kvh = hh >> 1;                        // G = NH/NKV = 2

  const u16* Qb = Q + (size_t)(b * NHQ + hh) * S_LEN * DHEAD;
  const u16* Kb = K + (size_t)(b * NKVH + kvh) * S_LEN * DHEAD;
  const u16* Vb = V + (size_t)(b * NKVH + kvh) * S_LEN * DHEAD;

  const int q0 = qt * 128;
  const int wq = q0 + w * 16;  // this wave's first q row

  // Q A-fragments (16 rows x 128 d), already RoPE'd and pre-scaled by 1/sqrt(D)
  F16 qf[4];
  {
    const u16* qrow = Qb + (size_t)(wq + ln) * DHEAD;
#pragma unroll
    for (int kc = 0; kc < 4; ++kc)
#pragma unroll
      for (int v = 0; v < 8; ++v)
        qf[kc].u[v] = *(const unsigned*)(qrow + kc * 32 + a_koff(half, v));
  }

  v8f acc[8];
#pragma unroll
  for (int i = 0; i < 8; ++i)
#pragma unroll
    for (int e = 0; e < 8; ++e) acc[i][e] = 0.0f;
  float mrow[8], lrow[8];
#pragma unroll
  for (int r = 0; r < 8; ++r) { mrow[r] = -1e30f; lrow[r] = 0.0f; }

  const int srow = t >> 2;         // staging: key row 0..63
  const int sseg = (t & 3) * 32;   // staging: d segment

  const int njt = 2 * qt + 2;      // causal: keys 0 .. q0+127
  for (int jt = 0; jt < njt; ++jt) {
    const int k0 = jt * 64;
    { // stage K tile row-major (serves as transposed-B for Q@K^T)
      const u16* g = Kb + (size_t)(k0 + srow) * DHEAD + sseg;
      u16* l = &Ks[srow * KSTR + sseg];
#if HAVE_ASYNC_LDS
      __builtin_amdgcn_global_load_async_to_lds_b128(TO_GLOBAL(g),      TO_LDS(l),      0, 0);
      __builtin_amdgcn_global_load_async_to_lds_b128(TO_GLOBAL(g + 8),  TO_LDS(l + 8),  0, 0);
      __builtin_amdgcn_global_load_async_to_lds_b128(TO_GLOBAL(g + 16), TO_LDS(l + 16), 0, 0);
      __builtin_amdgcn_global_load_async_to_lds_b128(TO_GLOBAL(g + 24), TO_LDS(l + 24), 0, 0);
#else
      const uint4* src = (const uint4*)g;
      uint4 a0 = src[0], a1 = src[1], a2 = src[2], a3 = src[3];
      uint4* dst = (uint4*)l;
      dst[0] = a0; dst[1] = a1; dst[2] = a2; dst[3] = a3;
#endif
    }
    { // stage V transposed: Vt[d][key]
      union { uint4 q; u16 s[8]; } v0, v1, v2, v3;
      const uint4* src = (const uint4*)(Vb + (size_t)(k0 + srow) * DHEAD + sseg);
      v0.q = src[0]; v1.q = src[1]; v2.q = src[2]; v3.q = src[3];
#pragma unroll
      for (int e = 0; e < 8; ++e) {
        Vt[(sseg + e) * VSTR + srow]      = v0.s[e];
        Vt[(sseg + 8 + e) * VSTR + srow]  = v1.s[e];
        Vt[(sseg + 16 + e) * VSTR + srow] = v2.s[e];
        Vt[(sseg + 24 + e) * VSTR + srow] = v3.s[e];
      }
    }
#if HAVE_ASYNC_LDS
    __builtin_amdgcn_s_wait_asynccnt(0);
#endif
    __syncthreads();

    // scores = Q(16x128) @ K^T(128x64) for this wave's rows
    v8f sc[4];
#pragma unroll
    for (int nt = 0; nt < 4; ++nt)
#pragma unroll
      for (int e = 0; e < 8; ++e) sc[nt][e] = 0.0f;
#pragma unroll
    for (int nt = 0; nt < 4; ++nt) {
#pragma unroll
      for (int kc = 0; kc < 4; ++kc) {
        F16 kf;
#pragma unroll
        for (int v = 0; v < 8; ++v)
          kf.u[v] = *(const unsigned*)&Ks[(nt * 16 + ln) * KSTR + kc * 32 + a_koff(half, v)];
        sc[nt] = __builtin_amdgcn_wmma_f32_16x16x32_bf16(
            false, qf[kc].v, false, kf.v, (short)0, sc[nt], false, false);
      }
    }

    // causal mask (C layout: M = r + 8*half, N = ln)
#pragma unroll
    for (int nt = 0; nt < 4; ++nt) {
      const int key = k0 + nt * 16 + ln;
#pragma unroll
      for (int r = 0; r < 8; ++r) {
        const int qi = wq + r + 8 * half;
        if (key > qi) sc[nt][r] = -1e30f;
      }
    }

    // online softmax per row (row values live in one 16-lane half)
#pragma unroll
    for (int r = 0; r < 8; ++r) {
      float tm = fmaxf(fmaxf(sc[0][r], sc[1][r]), fmaxf(sc[2][r], sc[3][r]));
#pragma unroll
      for (int msk = 1; msk < 16; msk <<= 1)
        tm = fmaxf(tm, __shfl_xor(tm, msk, 32));
      const float mnew  = fmaxf(mrow[r], tm);
      const float alpha = __expf(mrow[r] - mnew);
      mrow[r] = mnew;
      float rs = 0.0f;
#pragma unroll
      for (int nt = 0; nt < 4; ++nt) {
        const float pval = __expf(sc[nt][r] - mnew);
        sc[nt][r] = pval;
        rs += pval;
      }
#pragma unroll
      for (int msk = 1; msk < 16; msk <<= 1)
        rs += __shfl_xor(rs, msk, 32);
      lrow[r] = lrow[r] * alpha + rs;
#pragma unroll
      for (int d8 = 0; d8 < 8; ++d8) acc[d8][r] *= alpha;
    }

    // relayout P (C layout -> A layout) via wave-private LDS
#pragma unroll
    for (int nt = 0; nt < 4; ++nt)
#pragma unroll
      for (int r = 0; r < 8; ++r)
        Pb[w][(r + 8 * half) * PSTR + nt * 16 + ln] = f2bf(sc[nt][r]);
    asm volatile("s_wait_dscnt 0" ::: "memory");  // same-wave LDS RAW

    // O += P(16x64) @ V(64x128)
#pragma unroll
    for (int kc2 = 0; kc2 < 2; ++kc2) {
      F16 pf;
#pragma unroll
      for (int v = 0; v < 8; ++v)
        pf.u[v] = *(const unsigned*)&Pb[w][ln * PSTR + kc2 * 32 + a_koff(half, v)];
#pragma unroll
      for (int nt2 = 0; nt2 < 8; ++nt2) {
        F16 vf;
#pragma unroll
        for (int v = 0; v < 8; ++v)
          vf.u[v] = *(const unsigned*)&Vt[(nt2 * 16 + ln) * VSTR + kc2 * 32 + a_koff(half, v)];
        acc[nt2] = __builtin_amdgcn_wmma_f32_16x16x32_bf16(
            false, pf.v, false, vf.v, (short)0, acc[nt2], false, false);
      }
    }
    __syncthreads();
  }

  // normalize and store ctx as [b, s, hh*D + d] bf16 (token-major for Wo GEMM)
#pragma unroll
  for (int r = 0; r < 8; ++r) {
    const int qi = wq + r + 8 * half;
    const float inv = 1.0f / lrow[r];
    u16* dst = ctx + ((size_t)(b * S_LEN + qi)) * (NHQ * DHEAD) + hh * DHEAD;
#pragma unroll
    for (int nt2 = 0; nt2 < 8; ++nt2)
      dst[nt2 * 16 + ln] = f2bf(acc[nt2][r] * inv);
  }
}

// ---------------------------------------------------------------- launch
extern "C" void kernel_launch(void* const* d_in, const int* in_sizes, int n_in,
                              void* d_out, int out_size, void* d_ws, size_t ws_size,
                              hipStream_t stream) {
  (void)in_sizes; (void)n_in; (void)out_size; (void)ws_size;
  const float* hs = (const float*)d_in[0];
  const float* Wq = (const float*)d_in[1];
  const float* bq = (const float*)d_in[2];
  const float* Wk = (const float*)d_in[3];
  const float* bk = (const float*)d_in[4];
  const float* Wv = (const float*)d_in[5];
  const float* bv = (const float*)d_in[6];
  const float* Wo = (const float*)d_in[7];
  float* out = (float*)d_out;

  char* p = (char*)d_ws;
  const size_t nTok = (size_t)NB * S_LEN;  // 4096
  u16* Xb  = (u16*)p; p += nTok * HIDN * 2;
  u16* Wqb = (u16*)p; p += (size_t)HIDN * NHQ * DHEAD * 2;
  u16* Wkb = (u16*)p; p += (size_t)HIDN * NKVH * DHEAD * 2;
  u16* Wvb = (u16*)p; p += (size_t)HIDN * NKVH * DHEAD * 2;
  u16* Wob = (u16*)p; p += (size_t)NHQ * DHEAD * HIDN * 2;
  u16* Qw  = (u16*)p; p += nTok * NHQ * DHEAD * 2;
  u16* Kw  = (u16*)p; p += nTok * NKVH * DHEAD * 2;
  u16* Vw  = (u16*)p; p += nTok * NKVH * DHEAD * 2;
  u16* Cw  = (u16*)p; p += nTok * NHQ * DHEAD * 2;

  cvt_f32_to_bf16<<<2048, 256, 0, stream>>>(hs, Xb, (long)nTok * HIDN);
  cvt_f32_to_bf16<<<2048, 256, 0, stream>>>(Wq, Wqb, (long)HIDN * NHQ * DHEAD);
  cvt_f32_to_bf16<<<1024, 256, 0, stream>>>(Wk, Wkb, (long)HIDN * NKVH * DHEAD);
  cvt_f32_to_bf16<<<1024, 256, 0, stream>>>(Wv, Wvb, (long)HIDN * NKVH * DHEAD);
  cvt_f32_to_bf16<<<2048, 256, 0, stream>>>(Wo, Wob, (long)NHQ * DHEAD * HIDN);

  // QKV projections -> bf16 [B, h, S, D]
  gemm_bf16<<<dim3(16, 32), 256, 0, stream>>>(Xb, Wqb, bq, nullptr, Qw, 4096, 2048, 2048, 0, NHQ);
  gemm_bf16<<<dim3(8, 32),  256, 0, stream>>>(Xb, Wkb, bk, nullptr, Kw, 4096, 1024, 2048, 0, NKVH);
  gemm_bf16<<<dim3(8, 32),  256, 0, stream>>>(Xb, Wvb, bv, nullptr, Vw, 4096, 1024, 2048, 0, NKVH);

  // RoPE (Q pre-scaled by 1/sqrt(128))
  rope_bf16<<<4096, 256, 0, stream>>>(Qw, (long)NB * NHQ * S_LEN, 0.08838834764831845f);
  rope_bf16<<<2048, 256, 0, stream>>>(Kw, (long)NB * NKVH * S_LEN, 1.0f);

  // flash attention: grid = B * NH * (S/128) = 512
  attn_fwd<<<512, 256, 0, stream>>>(Qw, Kw, Vw, Cw);

  // output projection -> f32
  gemm_bf16<<<dim3(16, 32), 256, 0, stream>>>(Cw, Wob, nullptr, out, nullptr, 4096, 2048, 2048, 1, 0);
}
